// MHSA_72645076844837
// MI455X (gfx1250) — matrix operand order repaired and checked by reading, not compiled, not run on hardware
//
#include <hip/hip_runtime.h>
#include <math.h>

typedef __attribute__((ext_vector_type(16))) _Float16 v16h;
typedef __attribute__((ext_vector_type(8)))  float    v8f;

#define BATCH 2
#define SEQN  2048
#define DIM   256
#define HEADS 8
#define DH    32
#define FFN   1024
#define ROWS  (BATCH * SEQN)   // 4096

// ---------------------------------------------------------------- helpers

__device__ __forceinline__ float gelu_f(float x) {
  // exact gelu: 0.5*x*(1+erf(x/sqrt(2)))
  return 0.5f * x * (1.0f + erff(x * 0.70710678118654752f));
}

// DPP16 ROW_XMASK butterfly move within a 16-lane row (pure VALU, no LDS port)
template <int XORV>
__device__ __forceinline__ float dpp_xor16(float v) {
  int i = __builtin_bit_cast(int, v);
  int r = __builtin_amdgcn_update_dpp(0, i, 0x160 | XORV, 0xf, 0xf, true);
  return __builtin_bit_cast(float, r);
}

// reductions across the 16-lane half-wave that shares one C-matrix row
__device__ __forceinline__ float half16_max(float v) {
  v = fmaxf(v, dpp_xor16<1>(v));
  v = fmaxf(v, dpp_xor16<2>(v));
  v = fmaxf(v, dpp_xor16<4>(v));
  v = fmaxf(v, dpp_xor16<8>(v));
  return v;
}
__device__ __forceinline__ float half16_sum(float v) {
  v += dpp_xor16<1>(v);
  v += dpp_xor16<2>(v);
  v += dpp_xor16<4>(v);
  v += dpp_xor16<8>(v);
  return v;
}

// ---------------------------------------------------------------- layernorm 0
// wave-per-row: LN(X) -> f16 (Q branch input), raw X -> f16 (K/V branch input)
__global__ void ln0_prep_kernel(const float* __restrict__ X,
                                const float* __restrict__ g,
                                const float* __restrict__ b,
                                _Float16* __restrict__ Xh,
                                _Float16* __restrict__ Xnh) {
  int wave = threadIdx.x >> 5, lane = threadIdx.x & 31;
  int row = blockIdx.x * 8 + wave;
  if (row >= ROWS) return;
  const float* xr = X + (size_t)row * DIM;
  float v[8], s = 0.f, sq = 0.f;
#pragma unroll
  for (int i = 0; i < 8; ++i) { v[i] = xr[lane + 32 * i]; s += v[i]; sq += v[i] * v[i]; }
#pragma unroll
  for (int m = 1; m <= 16; m <<= 1) { s += __shfl_xor(s, m, 32); sq += __shfl_xor(sq, m, 32); }
  float mean = s * (1.0f / DIM);
  float var  = sq * (1.0f / DIM) - mean * mean;
  float rstd = rsqrtf(var + 1e-5f);
#pragma unroll
  for (int i = 0; i < 8; ++i) {
    int c = lane + 32 * i;
    float xn = (v[i] - mean) * rstd * g[c] + b[c];
    Xnh[(size_t)row * DIM + c] = (_Float16)xn;
    Xh [(size_t)row * DIM + c] = (_Float16)v[i];
  }
}

// ---------------------------------------------------------------- layernorm 1
__global__ void ln1_kernel(const float* __restrict__ Hres,
                           const float* __restrict__ g,
                           const float* __restrict__ b,
                           _Float16* __restrict__ out) {
  int wave = threadIdx.x >> 5, lane = threadIdx.x & 31;
  int row = blockIdx.x * 8 + wave;
  if (row >= ROWS) return;
  const float* xr = Hres + (size_t)row * DIM;
  float v[8], s = 0.f, sq = 0.f;
#pragma unroll
  for (int i = 0; i < 8; ++i) { v[i] = xr[lane + 32 * i]; s += v[i]; sq += v[i] * v[i]; }
#pragma unroll
  for (int m = 1; m <= 16; m <<= 1) { s += __shfl_xor(s, m, 32); sq += __shfl_xor(sq, m, 32); }
  float mean = s * (1.0f / DIM);
  float var  = sq * (1.0f / DIM) - mean * mean;
  float rstd = rsqrtf(var + 1e-5f);
#pragma unroll
  for (int i = 0; i < 8; ++i) {
    int c = lane + 32 * i;
    out[(size_t)row * DIM + c] = (_Float16)((v[i] - mean) * rstd * g[c] + b[c]);
  }
}

// ------------------------------------------------- f32 -> f16 with transpose
// in: f32 row-major [K][N]  ->  out: f16 [N][K] (so B fragments read contiguous)
__global__ void cvt_f16_T_kernel(const float* __restrict__ in, _Float16* __restrict__ out,
                                 int K, int N) {
  int i = blockIdx.x * blockDim.x + threadIdx.x;
  if (i < K * N) {
    int k = i / N, n = i - k * N;
    out[(size_t)n * K + k] = (_Float16)in[i];
  }
}

// ---------------------------------------------------------------- WMMA GEMM
// one wave per 16x16 C tile; A f16 row-major [M][K], Bt f16 TRANSPOSED [N][K].
// MODE 0: store f16        (K,V projections)
// MODE 1: store f32 + f16  (Q: residual + attention operand)
// MODE 2: gelu(x+bias)+resid -> f32            (head mix)
// MODE 3: gelu(gelu(x+bias)) -> f16            (ffn1)
// MODE 4: resid + gelu(x+bias) -> f32          (ffn2 + final residual)
template <int MODE>
__global__ void gemm_wmma_kernel(const _Float16* __restrict__ A,
                                 const _Float16* __restrict__ Bt,
                                 const float* __restrict__ bias,
                                 const float* __restrict__ resid,
                                 float* __restrict__ outF,
                                 _Float16* __restrict__ outH,
                                 int M, int Ncols, int K) {
  int wave = threadIdx.x >> 5, lane = threadIdx.x & 31;
  int hl = lane >> 4, ln = lane & 15;
  int tile_n = (blockIdx.x * 8 + wave) * 16;
  int tile_m = blockIdx.y * 16;
  if (tile_n >= Ncols || tile_m >= M) return;

  const _Float16* Arow = A  + (size_t)(tile_m + ln) * K;  // A frag: lane row m=ln
  const _Float16* Brow = Bt + (size_t)(tile_n + ln) * K;  // B frag: lane col n=ln (transposed)
  v8f c = {};
  for (int k0 = 0; k0 < K; k0 += 32) {
    if (k0 + 64 < K) {
      __builtin_prefetch(Arow + k0 + 64, 0, 1);
      __builtin_prefetch(Brow + k0 + 64, 0, 1);
    }
    v16h a, b;
#pragma unroll
    for (int v = 0; v < 8; ++v) {
      int ka = (v >> 2) * 16 + hl * 8 + (v & 3) * 2;      // 16-bit A 16x32 layout
      a[2 * v]     = Arow[k0 + ka];
      a[2 * v + 1] = Arow[k0 + ka + 1];
      int kb = (v >> 2) * 16 + hl * 8 + (v & 3) * 2;      // 16-bit B 32x16 layout
      b[2 * v]     = Brow[k0 + kb];                       // two 16B runs per lane
      b[2 * v + 1] = Brow[k0 + kb + 1];
    }
    c = __builtin_amdgcn_wmma_f32_16x16x32_f16(false, a, false, b, (short)0, c, false, false);
  }

  int coln = tile_n + ln;
#pragma unroll
  for (int r = 0; r < 8; ++r) {
    int rowm = tile_m + r + 8 * hl;                       // C layout: M = r + 8*(lane>>4)
    size_t idx = (size_t)rowm * Ncols + coln;
    float x = c[r];
    if (MODE == 0) {
      outH[idx] = (_Float16)x;
    } else if (MODE == 1) {
      outF[idx] = x;
      outH[idx] = (_Float16)x;
    } else if (MODE == 2) {
      outF[idx] = gelu_f(x + bias[coln]) + resid[idx];
    } else if (MODE == 3) {
      outH[idx] = (_Float16)gelu_f(gelu_f(x + bias[coln]));
    } else {
      outF[idx] = resid[idx] + gelu_f(x + bias[coln]);
    }
  }
}

// ---------------------------------------------------------------- attention
// one workgroup per (batch, head, 128 query rows); 8 waves x 16 rows each.
// Flash-attention online softmax; K/V staged in LDS by 128-key chunks.
__global__ void attn_kernel(const _Float16* __restrict__ Qh,
                            const _Float16* __restrict__ Kh,
                            const _Float16* __restrict__ Vh,
                            _Float16* __restrict__ Oh) {
  __shared__ _Float16 Ks[128][40];      // +8 half pad against bank conflicts
  __shared__ _Float16 Vs[128][40];
  __shared__ _Float16 Ps[8][16][36];    // per-wave P staging (C-layout -> A-layout)

  int qc = blockIdx.x & 15;
  int h  = (blockIdx.x >> 4) & 7;
  int b  = blockIdx.x >> 7;
  int wave = threadIdx.x >> 5, lane = threadIdx.x & 31;
  int hl = lane >> 4, ln = lane & 15;
  int qbase = qc * 128 + wave * 16;

  // q fragment (A layout 16x32 == 16 rows x full head dim), pre-scaled 1/sqrt(256)
  const _Float16* qp = Qh + (size_t)(b * SEQN + qbase + ln) * DIM + h * DH;
  v16h qf;
#pragma unroll
  for (int v = 0; v < 8; ++v) {
    int ka = (v >> 2) * 16 + hl * 8 + (v & 3) * 2;
    qf[2 * v]     = (_Float16)((float)qp[ka]     * 0.0625f);
    qf[2 * v + 1] = (_Float16)((float)qp[ka + 1] * 0.0625f);
  }

  float mrow[8], lrow[8];
#pragma unroll
  for (int r = 0; r < 8; ++r) { mrow[r] = -1e30f; lrow[r] = 0.f; }
  v8f o0 = {}, o1 = {};

  for (int kc = 0; kc < SEQN; kc += 128) {
    // cooperative chunk load: 128 keys x 32 halves for K and V (16 halves/thread each)
    {
      int row = threadIdx.x >> 1, seg = (threadIdx.x & 1) * 16;
      const _Float16* ks = Kh + (size_t)(b * SEQN + kc + row) * DIM + h * DH + seg;
      const _Float16* vs = Vh + (size_t)(b * SEQN + kc + row) * DIM + h * DH + seg;
#pragma unroll
      for (int j = 0; j < 16; ++j) { Ks[row][seg + j] = ks[j]; Vs[row][seg + j] = vs[j]; }
    }
    __syncthreads();

    for (int kt = 0; kt < 128; kt += 32) {
      // S tiles: q(16x32) @ K^T(32x16) for keys [kt,kt+16) and [kt+16,kt+32)
      v16h bk0, bk1;
#pragma unroll
      for (int v = 0; v < 8; ++v) {
        int kk = (v >> 2) * 16 + hl * 8 + (v & 3) * 2;
        bk0[2 * v]     = Ks[kt + ln][kk];
        bk0[2 * v + 1] = Ks[kt + ln][kk + 1];
        bk1[2 * v]     = Ks[kt + 16 + ln][kk];
        bk1[2 * v + 1] = Ks[kt + 16 + ln][kk + 1];
      }
      v8f zz = {};
      v8f s0 = __builtin_amdgcn_wmma_f32_16x16x32_f16(false, qf, false, bk0, (short)0, zz, false, false);
      v8f s1 = __builtin_amdgcn_wmma_f32_16x16x32_f16(false, qf, false, bk1, (short)0, zz, false, false);

      // online softmax per row (rows striped across 16-lane halves; DPP reductions)
#pragma unroll
      for (int r = 0; r < 8; ++r) {
        float a0 = s0[r], a1 = s1[r];
        float gm = half16_max(fmaxf(a0, a1));
        float mn = fmaxf(mrow[r], gm);
        float alpha = __expf(mrow[r] - mn);
        float p0 = __expf(a0 - mn), p1 = __expf(a1 - mn);
        lrow[r] = lrow[r] * alpha + half16_sum(p0 + p1);
        mrow[r] = mn;
        o0[r] *= alpha;
        o1[r] *= alpha;
        int prow = r + 8 * hl;
        Ps[wave][prow][ln]      = (_Float16)p0;
        Ps[wave][prow][16 + ln] = (_Float16)p1;
      }
      // same-wave LDS store -> load ordering
      asm volatile("s_wait_dscnt 0" ::: "memory");

      // reload P in A layout, build V B-fragments, accumulate O
      v16h pf, bv0, bv1;
#pragma unroll
      for (int v = 0; v < 8; ++v) {
        int ka = (v >> 2) * 16 + hl * 8 + (v & 3) * 2;
        pf[2 * v]     = Ps[wave][ln][ka];
        pf[2 * v + 1] = Ps[wave][ln][ka + 1];
        int kk = (v >> 2) * 16 + hl * 8 + (v & 3) * 2;
        bv0[2 * v]     = Vs[kt + kk][ln];
        bv0[2 * v + 1] = Vs[kt + kk + 1][ln];
        bv1[2 * v]     = Vs[kt + kk][16 + ln];
        bv1[2 * v + 1] = Vs[kt + kk + 1][16 + ln];
      }
      o0 = __builtin_amdgcn_wmma_f32_16x16x32_f16(false, pf, false, bv0, (short)0, o0, false, false);
      o1 = __builtin_amdgcn_wmma_f32_16x16x32_f16(false, pf, false, bv1, (short)0, o1, false, false);
    }
    __syncthreads();
  }

#pragma unroll
  for (int r = 0; r < 8; ++r) {
    float inv = 1.0f / lrow[r];
    int row = qbase + r + 8 * hl;
    size_t base = (size_t)(b * SEQN + row) * DIM + h * DH;
    Oh[base + ln]      = (_Float16)(o0[r] * inv);
    Oh[base + 16 + ln] = (_Float16)(o1[r] * inv);
  }
}

// ---------------------------------------------------------------- launch
extern "C" void kernel_launch(void* const* d_in, const int* in_sizes, int n_in,
                              void* d_out, int out_size, void* d_ws, size_t ws_size,
                              hipStream_t stream) {
  (void)in_sizes; (void)n_in; (void)out_size; (void)ws_size;
  const float* X    = (const float*)d_in[0];
  const float* Wq   = (const float*)d_in[1];
  const float* Wk   = (const float*)d_in[2];
  const float* Wv   = (const float*)d_in[3];
  const float* Wm   = (const float*)d_in[4];
  const float* bm   = (const float*)d_in[5];
  const float* W1   = (const float*)d_in[6];
  const float* bf1  = (const float*)d_in[7];
  const float* W2   = (const float*)d_in[8];
  const float* bf2  = (const float*)d_in[9];
  const float* ln0g = (const float*)d_in[10];
  const float* ln0b = (const float*)d_in[11];
  const float* ln1g = (const float*)d_in[12];
  const float* ln1b = (const float*)d_in[13];
  float* out = (float*)d_out;

  char* p = (char*)d_ws;
  auto alloc = [&](size_t bytes) -> char* {
    char* r = p;
    p += (bytes + 255) & ~(size_t)255;
    return r;
  };
  _Float16* Xh    = (_Float16*)alloc((size_t)ROWS * DIM * 2);
  _Float16* Xnh   = (_Float16*)alloc((size_t)ROWS * DIM * 2);
  float*    Qf    = (float*)   alloc((size_t)ROWS * DIM * 4);
  _Float16* Qh2   = (_Float16*)alloc((size_t)ROWS * DIM * 2);
  _Float16* Khh   = (_Float16*)alloc((size_t)ROWS * DIM * 2);
  _Float16* Vhh   = (_Float16*)alloc((size_t)ROWS * DIM * 2);
  _Float16* mhh   = (_Float16*)alloc((size_t)ROWS * DIM * 2);
  float*    Hres  = (float*)   alloc((size_t)ROWS * DIM * 4);
  _Float16* Hrffh = (_Float16*)alloc((size_t)ROWS * DIM * 2);
  _Float16* h1h   = (_Float16*)alloc((size_t)ROWS * FFN * 2);
  _Float16* WqhT  = (_Float16*)alloc((size_t)DIM * DIM * 2);
  _Float16* WkhT  = (_Float16*)alloc((size_t)DIM * DIM * 2);
  _Float16* WvhT  = (_Float16*)alloc((size_t)DIM * DIM * 2);
  _Float16* WmhT  = (_Float16*)alloc((size_t)DIM * DIM * 2);
  _Float16* W1hT  = (_Float16*)alloc((size_t)DIM * FFN * 2);
  _Float16* W2hT  = (_Float16*)alloc((size_t)FFN * DIM * 2);

  // stage 0: LN(X) and raw X to f16; weights to transposed f16
  ln0_prep_kernel<<<ROWS / 8, 256, 0, stream>>>(X, ln0g, ln0b, Xh, Xnh);
  cvt_f16_T_kernel<<<(DIM * DIM + 255) / 256, 256, 0, stream>>>(Wq, WqhT, DIM, DIM);
  cvt_f16_T_kernel<<<(DIM * DIM + 255) / 256, 256, 0, stream>>>(Wk, WkhT, DIM, DIM);
  cvt_f16_T_kernel<<<(DIM * DIM + 255) / 256, 256, 0, stream>>>(Wv, WvhT, DIM, DIM);
  cvt_f16_T_kernel<<<(DIM * DIM + 255) / 256, 256, 0, stream>>>(Wm, WmhT, DIM, DIM);
  cvt_f16_T_kernel<<<(DIM * FFN + 255) / 256, 256, 0, stream>>>(W1, W1hT, DIM, FFN);
  cvt_f16_T_kernel<<<(FFN * DIM + 255) / 256, 256, 0, stream>>>(W2, W2hT, FFN, DIM);

  // stage 1: Q/K/V projections
  dim3 g256(DIM / 16 / 8, ROWS / 16);   // (2, 256)
  gemm_wmma_kernel<1><<<g256, 256, 0, stream>>>(Xnh, WqhT, nullptr, nullptr, Qf, Qh2, ROWS, DIM, DIM);
  gemm_wmma_kernel<0><<<g256, 256, 0, stream>>>(Xh,  WkhT, nullptr, nullptr, nullptr, Khh, ROWS, DIM, DIM);
  gemm_wmma_kernel<0><<<g256, 256, 0, stream>>>(Xh,  WvhT, nullptr, nullptr, nullptr, Vhh, ROWS, DIM, DIM);

  // stage 2: flash attention per (batch, head, 128-row query block)
  attn_kernel<<<BATCH * HEADS * (SEQN / 128), 256, 0, stream>>>(Qh2, Khh, Vhh, mhh);

  // stage 3: head mix + gelu + residual(Q)
  gemm_wmma_kernel<2><<<g256, 256, 0, stream>>>(mhh, WmhT, bm, Qf, Hres, nullptr, ROWS, DIM, DIM);

  // stage 4: LN, FFN with fused double-gelu, final residual
  ln1_kernel<<<ROWS / 8, 256, 0, stream>>>(Hres, ln1g, ln1b, Hrffh);
  dim3 g1024(FFN / 16 / 8, ROWS / 16);  // (8, 256)
  gemm_wmma_kernel<3><<<g1024, 256, 0, stream>>>(Hrffh, W1hT, bf1, nullptr, nullptr, h1h, ROWS, FFN, DIM);
  gemm_wmma_kernel<4><<<g256, 256, 0, stream>>>(h1h, W2hT, bf2, Hres, out, nullptr, ROWS, DIM, FFN);
}
